// MoE_70179765617231
// MI455X (gfx1250) — matrix operand order repaired and checked by reading, not compiled, not run on hardware
//
#include <hip/hip_runtime.h>

#define DEVFN __device__ __forceinline__

typedef float        f32x4   __attribute__((ext_vector_type(4)));
typedef float        f32x8   __attribute__((ext_vector_type(8)));
typedef unsigned int u32x2   __attribute__((ext_vector_type(2)));
typedef unsigned int u32x4   __attribute__((ext_vector_type(4)));
typedef __bf16       bf16x16 __attribute__((ext_vector_type(16)));

union Frag { u32x4 u[2]; bf16x16 v; };

constexpr int Bc = 4, Tc = 2048, NT = Bc * Tc;   // 8192 tokens
constexpr int Dd = 1024, Hh = 2048;
constexpr int NE = 8, NS = 2;

DEVFN unsigned short f2bf(float x) {
  unsigned int u = __float_as_uint(x);
  u += 0x7fffu + ((u >> 16) & 1u);              // round-to-nearest-even
  return (unsigned short)(u >> 16);
}

DEVFN f32x8 zero8() {
  f32x8 z;
#pragma unroll
  for (int i = 0; i < 8; i++) z[i] = 0.0f;
  return z;
}

// ---------------- zero / convert kernels ----------------

__global__ void zero_f32_k(float* p, int n4) {
  int i = blockIdx.x * blockDim.x + threadIdx.x;
  if (i < n4) {
    u32x4 z; z[0] = 0u; z[1] = 0u; z[2] = 0u; z[3] = 0u;
    ((u32x4*)p)[i] = z;
  }
}

__global__ void zero_pos_k(int* pos) {
  if (threadIdx.x < 16) pos[threadIdx.x] = 0;
}

__global__ void cvt_bf16_k(const float* __restrict__ in,
                           unsigned short* __restrict__ out, int n4) {
  int i = blockIdx.x * blockDim.x + threadIdx.x;
  if (i >= n4) return;
  f32x4 v = ((const f32x4*)in)[i];
  u32x2 r;
  r.x = (unsigned)f2bf(v.x) | ((unsigned)f2bf(v.y) << 16);
  r.y = (unsigned)f2bf(v.z) | ((unsigned)f2bf(v.w) << 16);
  ((u32x2*)out)[i] = r;
}

// Batched transpose + f32->bf16: in (batch,R,C) -> out (batch,C,R)
__global__ void transpose_bf16_k(const float* __restrict__ in,
                                 unsigned short* __restrict__ out,
                                 int R, int C) {
  __shared__ float t[32][33];
  int b = blockIdx.z;
  int cb = blockIdx.x * 32, rb = blockIdx.y * 32;
  const float* ip = in + (size_t)b * R * C;
  unsigned short* op = out + (size_t)b * R * C;
  int tx = threadIdx.x, ty = threadIdx.y;     // block (32,8)
#pragma unroll
  for (int j = 0; j < 4; j++)
    t[ty + j * 8][tx] = ip[(size_t)(rb + ty + j * 8) * C + cb + tx];
  __syncthreads();
#pragma unroll
  for (int j = 0; j < 4; j++)
    op[(size_t)(cb + ty + j * 8) * R + rb + tx] = f2bf(t[tx][ty + j * 8]);
}

// ---------------- gate: softmax + top-2 ----------------

__global__ void gate_k(const float* __restrict__ x,
                       const float* __restrict__ gw,
                       const float* __restrict__ gb,
                       int* __restrict__ gidx, float* __restrict__ gwt) {
  int tok  = (blockIdx.x * blockDim.x + threadIdx.x) >> 5;
  int lane = threadIdx.x & 31;
  if (tok >= NT) return;
  const float* xr = x + (size_t)tok * Dd;
  float acc[8];
#pragma unroll
  for (int e = 0; e < 8; e++) acc[e] = 0.0f;
  for (int d = lane; d < Dd; d += 32) {
    float xv = xr[d];
    f32x4 ga  = ((const f32x4*)(gw + d * 8))[0];
    f32x4 gbv = ((const f32x4*)(gw + d * 8))[1];
    acc[0] += xv * ga.x;  acc[1] += xv * ga.y;
    acc[2] += xv * ga.z;  acc[3] += xv * ga.w;
    acc[4] += xv * gbv.x; acc[5] += xv * gbv.y;
    acc[6] += xv * gbv.z; acc[7] += xv * gbv.w;
  }
#pragma unroll
  for (int e = 0; e < 8; e++)
#pragma unroll
    for (int off = 16; off; off >>= 1)
      acc[e] += __shfl_xor(acc[e], off, 32);
  if (lane == 0) {
    float mx = -1e30f;
#pragma unroll
    for (int e = 0; e < 8; e++) { acc[e] += gb[e]; mx = fmaxf(mx, acc[e]); }
    float s = 0.0f, p[8];
#pragma unroll
    for (int e = 0; e < 8; e++) { p[e] = __expf(acc[e] - mx); s += p[e]; }
    float inv = 1.0f / s;
#pragma unroll
    for (int e = 0; e < 8; e++) p[e] *= inv;
    int i0 = 0;
#pragma unroll
    for (int e = 1; e < 8; e++) if (p[e] > p[i0]) i0 = e;
    int i1 = (i0 == 0) ? 1 : 0;
#pragma unroll
    for (int e = 0; e < 8; e++) if (e != i0 && e != i1 && p[e] > p[i1]) i1 = e;
    gidx[tok * 2 + 0] = i0; gidx[tok * 2 + 1] = i1;
    gwt[tok * 2 + 0]  = p[i0]; gwt[tok * 2 + 1] = p[i1];
  }
}

__global__ void scatter_k(const int* __restrict__ gidx,
                          const float* __restrict__ gwt,
                          int* __restrict__ toklist, float* __restrict__ wlist,
                          int* __restrict__ pos) {
  int t = blockIdx.x * blockDim.x + threadIdx.x;
  if (t >= NT) return;
#pragma unroll
  for (int k = 0; k < 2; k++) {
    int e = gidx[t * 2 + k];
    int p = atomicAdd(&pos[e], 1);
    toklist[e * NT + p] = t;
    wlist[e * NT + p]   = gwt[t * 2 + k];
  }
}

// ---------------- fused grouped SwiGLU expert FFN ----------------
// grid: (NT/32, NE+NS); block 256 (8 waves). Weights pre-transposed:
// w1t/w2t: (H,D) bf16 (K-contiguous for GEMM1 B), w3t: (D,H) bf16.

__global__ __launch_bounds__(256)
void expert_k(const unsigned short* __restrict__ xb,
              const unsigned short* __restrict__ rw1t,
              const unsigned short* __restrict__ rw2t,
              const unsigned short* __restrict__ rw3t,
              const float* __restrict__ rb1, const float* __restrict__ rb2,
              const float* __restrict__ rb3,
              const unsigned short* __restrict__ sw1t,
              const unsigned short* __restrict__ sw2t,
              const unsigned short* __restrict__ sw3t,
              const float* __restrict__ sb1, const float* __restrict__ sb2,
              const float* __restrict__ sb3,
              const int* __restrict__ toklist, const float* __restrict__ wlist,
              const int* __restrict__ pos, float* __restrict__ out) {
  __shared__ __align__(16) unsigned short xs[32 * 1032];   // 32 x 1024 bf16, padded
  __shared__ __align__(16) float p1s[32 * 68];
  __shared__ __align__(16) float p2s[32 * 68];
  __shared__ __align__(16) unsigned short glus[32 * 72];   // 32 x 64 bf16, padded
  __shared__ int   tok_s[32];
  __shared__ float w_s[32];

  const int e     = blockIdx.y;
  const int mbase = blockIdx.x * 32;
  const int tid   = threadIdx.x;
  const int lane  = tid & 31;
  const int wv    = tid >> 5;
  const int half  = lane >> 4;
  const int l15   = lane & 15;

  const unsigned short *w1, *w2, *w3;
  const float *b1, *b2, *b3;
  int count;
  if (e < NE) {
    count = pos[e];
    if (mbase >= count) return;                 // uniform across block
    w1 = rw1t + (size_t)e * Hh * Dd;
    w2 = rw2t + (size_t)e * Hh * Dd;
    w3 = rw3t + (size_t)e * Dd * Hh;
    b1 = rb1 + e * Hh; b2 = rb2 + e * Hh; b3 = rb3 + e * Dd;
  } else {
    int se = e - NE;
    count = NT;
    w1 = sw1t + (size_t)se * Hh * Dd;
    w2 = sw2t + (size_t)se * Hh * Dd;
    w3 = sw3t + (size_t)se * Dd * Hh;
    b1 = sb1 + se * Hh; b2 = sb2 + se * Hh; b3 = sb3 + se * Dd;
  }

  if (tid < 32) {
    int m = mbase + tid;
    int tk = 0; float wt = 0.0f;
    if (e < NE) {
      if (m < count) { tk = toklist[e * NT + m]; wt = wlist[e * NT + m]; }
    } else { tk = m; wt = 0.5f; }
    tok_s[tid] = tk; w_s[tid] = wt;
  }
  __syncthreads();

  // gather 32 token rows of bf16 x into LDS (16B chunks)
  for (int c = tid; c < 32 * 128; c += 256) {
    int r = c >> 7, cc = c & 127;
    u32x4 v = *(const u32x4*)(xb + (size_t)tok_s[r] * Dd + cc * 8);
    *(u32x4*)(&xs[r * 1032 + cc * 8]) = v;
  }
  __syncthreads();

  const bool isP1 = (wv < 4);
  const int  nt4  = wv & 3;
  const unsigned short* wp = isP1 ? w1 : w2;
  const float* bp = isP1 ? b1 : b2;

  f32x8 C[2][8];
#pragma unroll
  for (int mt = 0; mt < 2; mt++)
#pragma unroll
    for (int nt = 0; nt < 8; nt++) C[mt][nt] = zero8();

  for (int hc = 0; hc < Hh; hc += 64) {
    // ---- GEMM1: p = x @ w  (this wave: one 16-col slab of p1 or p2) ----
    f32x8 p[2]; p[0] = zero8(); p[1] = zero8();
    const int colbase = hc + nt4 * 16;
    const unsigned short* wrow = wp + (size_t)(colbase + l15) * Dd + half * 16;

    // prefetch next H-chunk's W1/W2 row into caches (global_prefetch_b8)
    if (hc + 64 < Hh)
      __builtin_prefetch((const void*)(wrow + (size_t)64 * Dd), 0, 0);

    // double-buffered B fragments: prefetch next K-step before consuming
    Frag fb[2];
    fb[0].u[0] = *(const u32x4*)(wrow);
    fb[0].u[1] = *(const u32x4*)(wrow + 8);
#pragma unroll 4
    for (int kb = 0; kb < Dd; kb += 32) {
      const int cur = (kb >> 5) & 1;
      const int nxt = cur ^ 1;
      if (kb + 32 < Dd) {
        fb[nxt].u[0] = *(const u32x4*)(wrow + kb + 32);
        fb[nxt].u[1] = *(const u32x4*)(wrow + kb + 40);
      }
#pragma unroll
      for (int mt = 0; mt < 2; mt++) {
        const unsigned short* ap = &xs[(mt * 16 + l15) * 1032 + kb + half * 16];
        Frag fa;
        fa.u[0] = *(const u32x4*)ap;
        fa.u[1] = *(const u32x4*)(ap + 8);
        p[mt] = __builtin_amdgcn_wmma_f32_16x16x32_bf16(
            false, fa.v, false, fb[cur].v, (short)0, p[mt], false, false);
      }
    }
    {
      float bias = bp[colbase + l15];
      float* pb = isP1 ? p1s : p2s;
#pragma unroll
      for (int mt = 0; mt < 2; mt++)
#pragma unroll
        for (int i = 0; i < 8; i++) {
          float v = p[mt][i] + bias;
          if (isP1) v = v / (1.0f + __expf(-v));   // silu
          pb[(mt * 16 + i + half * 8) * 68 + nt4 * 16 + l15] = v;
        }
    }
    __syncthreads();
    // ---- glu = silu(p1) * p2 -> bf16 tile ----
#pragma unroll
    for (int j = 0; j < 8; j++) {
      int id = tid + j * 256;
      int r = id >> 6, c = id & 63;
      glus[r * 72 + c] = f2bf(p1s[r * 68 + c] * p2s[r * 68 + c]);
    }
    __syncthreads();
    // ---- GEMM2: C += glu @ w3  (this wave owns 128 output cols) ----
    const unsigned short* w3base =
        w3 + (size_t)(wv * 128 + l15) * Hh + hc + half * 16;
#pragma unroll
    for (int ks = 0; ks < 2; ks++) {
      Frag fa[2];
#pragma unroll
      for (int mt = 0; mt < 2; mt++) {
        const unsigned short* ap = &glus[(mt * 16 + l15) * 72 + ks * 32 + half * 16];
        fa[mt].u[0] = *(const u32x4*)ap;
        fa[mt].u[1] = *(const u32x4*)(ap + 8);
      }
      // double-buffered B fragments across the 8 N-tiles
      Frag fbb[2];
      fbb[0].u[0] = *(const u32x4*)(w3base + ks * 32);
      fbb[0].u[1] = *(const u32x4*)(w3base + ks * 32 + 8);
#pragma unroll
      for (int nt = 0; nt < 8; nt++) {
        const int cur = nt & 1;
        const int nxt = cur ^ 1;
        if (nt < 7) {
          const unsigned short* bg = w3base + (size_t)(nt + 1) * 16 * Hh + ks * 32;
          fbb[nxt].u[0] = *(const u32x4*)(bg);
          fbb[nxt].u[1] = *(const u32x4*)(bg + 8);
        }
#pragma unroll
        for (int mt = 0; mt < 2; mt++)
          C[mt][nt] = __builtin_amdgcn_wmma_f32_16x16x32_bf16(
              false, fa[mt].v, false, fbb[cur].v, (short)0, C[mt][nt], false, false);
      }
    }
    __syncthreads();
  }

  // ---- epilogue: weight * (C + b3) -> atomic add into out ----
  float bias[8];
#pragma unroll
  for (int nt = 0; nt < 8; nt++) bias[nt] = b3[wv * 128 + nt * 16 + l15];
#pragma unroll
  for (int mt = 0; mt < 2; mt++)
#pragma unroll
    for (int i = 0; i < 8; i++) {
      int row = mt * 16 + i + half * 8;
      float wt = w_s[row];
      if (wt != 0.0f) {
        float* orow = out + (size_t)tok_s[row] * Dd;
#pragma unroll
        for (int nt = 0; nt < 8; nt++) {
          int col = wv * 128 + nt * 16 + l15;
          __hip_atomic_fetch_add(orow + col, wt * (C[mt][nt][i] + bias[nt]),
                                 __ATOMIC_RELAXED, __HIP_MEMORY_SCOPE_AGENT);
        }
      }
    }
}

// ---------------- host launcher ----------------

extern "C" void kernel_launch(void* const* d_in, const int* in_sizes, int n_in,
                              void* d_out, int out_size, void* d_ws, size_t ws_size,
                              hipStream_t stream) {
  (void)in_sizes; (void)n_in; (void)out_size; (void)ws_size;
  const float* x      = (const float*)d_in[0];
  const float* gate_w = (const float*)d_in[1];
  const float* gate_b = (const float*)d_in[2];
  const float* rw1    = (const float*)d_in[3];
  const float* rb1    = (const float*)d_in[4];
  const float* rw2    = (const float*)d_in[5];
  const float* rb2    = (const float*)d_in[6];
  const float* rw3    = (const float*)d_in[7];
  const float* rb3    = (const float*)d_in[8];
  const float* sw1    = (const float*)d_in[9];
  const float* sb1    = (const float*)d_in[10];
  const float* sw2    = (const float*)d_in[11];
  const float* sb2    = (const float*)d_in[12];
  const float* sw3    = (const float*)d_in[13];
  const float* sb3    = (const float*)d_in[14];
  float* out = (float*)d_out;

  char* ws = (char*)d_ws;
  size_t off = 0;
  auto alloc = [&](size_t bytes) {
    size_t cur = off;
    off += (bytes + 255) & ~(size_t)255;
    return cur;
  };
  unsigned short* xb   = (unsigned short*)(ws + alloc((size_t)NT * Dd * 2));
  unsigned short* rw1t = (unsigned short*)(ws + alloc((size_t)NE * Hh * Dd * 2));
  unsigned short* rw2t = (unsigned short*)(ws + alloc((size_t)NE * Hh * Dd * 2));
  unsigned short* rw3t = (unsigned short*)(ws + alloc((size_t)NE * Dd * Hh * 2));
  unsigned short* sw1t = (unsigned short*)(ws + alloc((size_t)NS * Hh * Dd * 2));
  unsigned short* sw2t = (unsigned short*)(ws + alloc((size_t)NS * Hh * Dd * 2));
  unsigned short* sw3t = (unsigned short*)(ws + alloc((size_t)NS * Dd * Hh * 2));
  int*   gidx    = (int*)  (ws + alloc((size_t)NT * 2 * 4));
  float* gwt     = (float*)(ws + alloc((size_t)NT * 2 * 4));
  int*   toklist = (int*)  (ws + alloc((size_t)NE * NT * 4));
  float* wlist   = (float*)(ws + alloc((size_t)NE * NT * 4));
  int*   pos     = (int*)  (ws + alloc(64));

  // 1) zero output + expert counters
  zero_f32_k<<<(NT * Dd / 4 + 255) / 256, 256, 0, stream>>>(out, NT * Dd / 4);
  zero_pos_k<<<1, 32, 0, stream>>>(pos);

  // 2) convert x, transpose-convert weights to bf16
  cvt_bf16_k<<<(NT * Dd / 4 + 255) / 256, 256, 0, stream>>>(x, xb, NT * Dd / 4);
  dim3 tb(32, 8);
  transpose_bf16_k<<<dim3(Hh / 32, Dd / 32, NE), tb, 0, stream>>>(rw1, rw1t, Dd, Hh);
  transpose_bf16_k<<<dim3(Hh / 32, Dd / 32, NE), tb, 0, stream>>>(rw2, rw2t, Dd, Hh);
  transpose_bf16_k<<<dim3(Dd / 32, Hh / 32, NE), tb, 0, stream>>>(rw3, rw3t, Hh, Dd);
  transpose_bf16_k<<<dim3(Hh / 32, Dd / 32, NS), tb, 0, stream>>>(sw1, sw1t, Dd, Hh);
  transpose_bf16_k<<<dim3(Hh / 32, Dd / 32, NS), tb, 0, stream>>>(sw2, sw2t, Dd, Hh);
  transpose_bf16_k<<<dim3(Dd / 32, Hh / 32, NS), tb, 0, stream>>>(sw3, sw3t, Hh, Dd);

  // 3) gate + scatter
  gate_k<<<NT / 8, 256, 0, stream>>>(x, gate_w, gate_b, gidx, gwt);
  scatter_k<<<NT / 256, 256, 0, stream>>>(gidx, gwt, toklist, wlist, pos);

  // 4) fused grouped expert FFN (routed experts 0..7, shared 8..9)
  expert_k<<<dim3(NT / 32, NE + NS), 256, 0, stream>>>(
      xb, rw1t, rw2t, rw3t, rb1, rb2, rb3,
      sw1t, sw2t, sw3t, sb1, sb2, sb3,
      toklist, wlist, pos, out);
}
